// AttnBlock2D_10204842295720
// MI455X (gfx1250) — matrix operand order repaired and checked by reading, not compiled, not run on hardware
//
#include <hip/hip_runtime.h>
#include <hip/hip_bf16.h>
#include <math.h>

#define B_   8
#define C_   512
#define HW_  4096
#define NG   32
#define CPG  16

typedef __bf16 bf16;
typedef __attribute__((ext_vector_type(16))) __bf16 v16bf;
typedef __attribute__((ext_vector_type(8)))  __bf16 v8bf;
typedef __attribute__((ext_vector_type(8)))  float  v8f;
typedef __attribute__((ext_vector_type(4)))  int    v4i;
typedef __attribute__((address_space(1))) v4i g_v4i;   // global int4
typedef __attribute__((address_space(3))) v4i l_v4i;   // LDS int4

// ---------------------------------------------------------------------------
// CDNA5 async global->LDS copy (ASYNCcnt path), guarded with fallback.
// Builtin signature (from hipcc diagnostic): (global v4i*, lds v4i*, imm, imm)
// ---------------------------------------------------------------------------
#if defined(__has_builtin)
#if __has_builtin(__builtin_amdgcn_global_load_async_to_lds_b128) && \
    __has_builtin(__builtin_amdgcn_s_wait_asynccnt)
#define HAS_ASYNC_LDS 1
#endif
#endif

static __device__ inline void cp_async16(bf16* ldst, const bf16* gsrc) {
#ifdef HAS_ASYNC_LDS
  __builtin_amdgcn_global_load_async_to_lds_b128(
      (g_v4i*)gsrc, (l_v4i*)ldst, 0, 0);
#else
  *(v8bf*)ldst = *(const v8bf*)gsrc;
#endif
}
static __device__ inline void cp_async_wait() {
#ifdef HAS_ASYNC_LDS
  __builtin_amdgcn_s_wait_asynccnt(0);
#endif
}

// ---------------------------------------------------------------------------
// WMMA fragment helpers (wave32, V_WMMA_F32_16X16X32_BF16; A/B use contiguous k, not ISA table order)
// A (16x32 bf16): lane L -> row M=L&15; elements t = K(half*16 + t),
//                 half = L>>4.
// B (32x16 bf16): lane L -> col N=L&15; elements t = K(half*16 + t).
// C (16x16 f32):  lane L -> col N=L&15; VGPR r -> row M = r + 8*half.
// ---------------------------------------------------------------------------
static __device__ inline v16bf make_a_frag(const bf16* base, int ld, int row0,
                                           int col0, int lane) {
  int m = lane & 15, h = lane >> 4;
  const bf16* p = base + (size_t)(row0 + m) * ld + col0;
  v8bf lo = *(const v8bf*)(p + h * 16);
  v8bf hi = *(const v8bf*)(p + h * 16 + 8);
  v16bf a;
#pragma unroll
  for (int t = 0; t < 8; ++t) { a[t] = lo[t]; a[t + 8] = hi[t]; }
  return a;
}

static __device__ inline v16bf make_b_frag(const bf16* base, int ld, int row0,
                                           int col0, int lane) {
  int n = lane & 15, h = lane >> 4;
  const bf16* p = base + (size_t)(row0 + n) * ld + col0 + h * 16;
  v8bf lo = *(const v8bf*)(p);
  v8bf hi = *(const v8bf*)(p + 8);
  v16bf b;
#pragma unroll
  for (int t = 0; t < 8; ++t) { b[t] = lo[t]; b[t + 8] = hi[t]; }
  return b;
}

static __device__ inline v8f wmma_bf16(v16bf a, v16bf b, v8f c) {
  return __builtin_amdgcn_wmma_f32_16x16x32_bf16(false, a, false, b,
                                                 (short)0, c, false, false);
}

// ---------------------------------------------------------------------------
// Kernel 1: fp32 weights -> bf16
// ---------------------------------------------------------------------------
__global__ void k_w2bf(const float* __restrict__ w, bf16* __restrict__ o, int n) {
  int i = blockIdx.x * blockDim.x + threadIdx.x;
  if (i < n) o[i] = (bf16)w[i];
}

// ---------------------------------------------------------------------------
// Kernel 2: GroupNorm statistics. One block per (b, group); group channels are
// contiguous -> coalesced flat reduction over 16*4096 floats.
// ---------------------------------------------------------------------------
__global__ void k_gnstats(const float* __restrict__ x, float* __restrict__ stats) {
  int blk = blockIdx.x;               // b*32 + g
  int b = blk >> 5, g = blk & 31;
  const float* xp = x + ((size_t)b * C_ + (size_t)g * CPG) * HW_;
  float s = 0.f, s2 = 0.f;
  for (int i = threadIdx.x; i < CPG * HW_; i += 256) {
    float v = xp[i];
    s += v; s2 += v * v;
  }
  __shared__ float rs[256], rs2[256];
  rs[threadIdx.x] = s; rs2[threadIdx.x] = s2;
  __syncthreads();
  for (int off = 128; off > 0; off >>= 1) {
    if (threadIdx.x < off) {
      rs[threadIdx.x]  += rs[threadIdx.x + off];
      rs2[threadIdx.x] += rs2[threadIdx.x + off];
    }
    __syncthreads();
  }
  if (threadIdx.x == 0) {
    const float inv = 1.0f / (float)(CPG * HW_);
    float mean = rs[0] * inv;
    float var  = rs2[0] * inv - mean * mean;
    stats[blk * 2 + 0] = mean;
    stats[blk * 2 + 1] = rsqrtf(var + 1e-5f);
  }
}

// ---------------------------------------------------------------------------
// Kernel 3: GroupNorm affine + transpose [b][c][s] -> Hn [b][s][c] bf16
// via 32x32 LDS tile (coalesced both directions).
// ---------------------------------------------------------------------------
__global__ void k_gnapply(const float* __restrict__ x, const float* __restrict__ stats,
                          const float* __restrict__ sc, const float* __restrict__ bi,
                          bf16* __restrict__ hn) {
  __shared__ float tile[32][33];
  int s0 = blockIdx.x * 32, c0 = blockIdx.y * 32, b = blockIdx.z;
  int sl = threadIdx.x & 31, cl = threadIdx.x >> 5;
#pragma unroll
  for (int p = 0; p < 4; ++p) {
    int clp = cl + p * 8;
    int c = c0 + clp;
    int gg = b * NG + (c >> 4);
    float mean = stats[gg * 2 + 0], rstd = stats[gg * 2 + 1];
    float v = x[((size_t)b * C_ + c) * HW_ + s0 + sl];
    tile[clp][sl] = (v - mean) * rstd * sc[c] + bi[c];
  }
  __syncthreads();
  int cl2 = threadIdx.x & 31, sl2 = threadIdx.x >> 5;
  bf16* hp = hn + (size_t)b * HW_ * C_;
#pragma unroll
  for (int p = 0; p < 4; ++p) {
    int slp = sl2 + p * 8;
    hp[(size_t)(s0 + slp) * C_ + c0 + cl2] = (bf16)tile[cl2][slp];
  }
}

// ---------------------------------------------------------------------------
// Kernel 4/6: bf16 GEMM  Y[o][s] = sum_c W[o][c] * H[b][s][c]
// 128x128 block tile, 8 waves (2x4), each wave a 64x32 sub-tile (4x2 C frags).
// DOUBLE-BUFFERED LDS staging (async-to-LDS when available): tile for step+1
// streams in while WMMA consumes step's tile. LDT=40 pads vs bank conflicts.
// mode 0: Q -> [b][s][o] bf16 *qscale | 1: K -> [b][s][o] | 2: V -> [b][o][s]
// mode 3: proj -> fp32 [b][o][s] = x + acc + bias (residual fused)
// ---------------------------------------------------------------------------
#define LDT 40
static __device__ inline void stage_tiles(bf16* la, bf16* lb,
                                          const bf16* __restrict__ A,
                                          const bf16* __restrict__ Bp,
                                          int o0, int s0, int kk, int tid) {
#pragma unroll
  for (int p = 0; p < 2; ++p) {
    int q = tid + p * 256;
    int row = q >> 2, ch = q & 3;
    cp_async16(&la[row * LDT + ch * 8], &A[(size_t)(o0 + row) * C_ + kk + ch * 8]);
    cp_async16(&lb[row * LDT + ch * 8], &Bp[(size_t)(s0 + row) * C_ + kk + ch * 8]);
  }
}

__global__ void k_gemm(const bf16* __restrict__ A, const bf16* __restrict__ Bsrc,
                       const float* __restrict__ bias, const float* __restrict__ xres,
                       bf16* __restrict__ outb, float* __restrict__ outf,
                       int mode, float qscale) {
  __shared__ __align__(16) bf16 lA[2][128 * LDT];
  __shared__ __align__(16) bf16 lB[2][128 * LDT];
  int b  = blockIdx.z;
  int o0 = blockIdx.y * 128;
  int s0 = blockIdx.x * 128;
  const bf16* Bp = Bsrc + (size_t)b * HW_ * C_;
  int tid = threadIdx.x;
  int lane = tid & 31, w = tid >> 5;
  int wm = w >> 2, wn = w & 3;
  int mbase = wm * 64, nbase = wn * 32;

  v8f acc[4][2];
#pragma unroll
  for (int i = 0; i < 4; ++i)
#pragma unroll
    for (int j = 0; j < 2; ++j)
#pragma unroll
      for (int r = 0; r < 8; ++r) acc[i][j][r] = 0.f;

  stage_tiles(lA[0], lB[0], A, Bp, o0, s0, 0, tid);
  cp_async_wait();
  __syncthreads();

  const int NSTEP = C_ / 32;   // 16
  for (int step = 0; step < NSTEP; ++step) {
    int cur = step & 1;
    if (step + 1 < NSTEP)
      stage_tiles(lA[cur ^ 1], lB[cur ^ 1], A, Bp, o0, s0, (step + 1) * 32, tid);

    v16bf af[4], bfr[2];
#pragma unroll
    for (int i = 0; i < 4; ++i) af[i]  = make_a_frag(lA[cur], LDT, mbase + i * 16, 0, lane);
#pragma unroll
    for (int j = 0; j < 2; ++j) bfr[j] = make_b_frag(lB[cur], LDT, nbase + j * 16, 0, lane);
#pragma unroll
    for (int i = 0; i < 4; ++i)
#pragma unroll
      for (int j = 0; j < 2; ++j)
        acc[i][j] = wmma_bf16(af[i], bfr[j], acc[i][j]);

    cp_async_wait();
    __syncthreads();
  }

  int half = lane >> 4, nl = lane & 15;
  v8f bv[4];
#pragma unroll
  for (int i = 0; i < 4; ++i)
    bv[i] = *(const v8f*)&bias[o0 + mbase + i * 16 + 8 * half];

#pragma unroll
  for (int i = 0; i < 4; ++i)
#pragma unroll
    for (int j = 0; j < 2; ++j)
#pragma unroll
      for (int r = 0; r < 8; ++r) {
        int o = o0 + mbase + i * 16 + r + 8 * half;
        int s = s0 + nbase + j * 16 + nl;
        float v = acc[i][j][r] + bv[i][r];
        if (mode == 0) {
          outb[((size_t)b * HW_ + s) * C_ + o] = (bf16)(v * qscale);
        } else if (mode == 1) {
          outb[((size_t)b * HW_ + s) * C_ + o] = (bf16)v;
        } else if (mode == 2) {
          outb[((size_t)b * C_ + o) * HW_ + s] = (bf16)v;
        } else {
          size_t idx = ((size_t)b * C_ + o) * HW_ + s;
          outf[idx] = xres[idx] + v;
        }
      }
}

// ---------------------------------------------------------------------------
// Kernel 5: flash attention. One block = 32 query rows, 8 waves.
// S GEMM: waves split K(=c,512) 8 ways; partials go to per-wave LDS slabs
// (no atomics, no zeroing); softmax sums the 8 slabs. PV GEMM: wave w owns
// channel chunk [w*64,w*64+64). K/V fragments are loaded straight from global
// (per-batch K+V = 8MB -> L2-resident); V frags prefetched at iteration start
// so their latency hides under the S GEMM + softmax barrier.
// ---------------------------------------------------------------------------
#define PLD 40
__global__ void k_flash(const bf16* __restrict__ Qg, const bf16* __restrict__ Kg,
                        const bf16* __restrict__ Vg, bf16* __restrict__ Og) {
  __shared__ float Sp[8][32 * 33];                 // per-wave score partials
  __shared__ __align__(16) bf16 Ps[32 * PLD];
  __shared__ float mrow[32], lrow[32], arow[32];

  int b  = blockIdx.y;
  int q0 = blockIdx.x * 32;
  int tid = threadIdx.x, lane = tid & 31, w = tid >> 5;
  int half = lane >> 4, nl = lane & 15;
  const bf16* Qp = Qg + (size_t)b * HW_ * C_;
  const bf16* Kp = Kg + (size_t)b * HW_ * C_;
  const bf16* Vp = Vg + (size_t)b * C_ * HW_;

  // Q fragments for this wave's c-chunk [w*64, w*64+64), loaded once.
  v16bf qf[2][2];
#pragma unroll
  for (int i = 0; i < 2; ++i)
#pragma unroll
    for (int st = 0; st < 2; ++st)
      qf[i][st] = make_a_frag(Qp, C_, q0 + i * 16, w * 64 + st * 32, lane);

  v8f oacc[2][4];
#pragma unroll
  for (int i = 0; i < 2; ++i)
#pragma unroll
    for (int j = 0; j < 4; ++j)
#pragma unroll
      for (int r = 0; r < 8; ++r) oacc[i][j][r] = 0.f;

  if (tid < 32) { mrow[tid] = -3.0e38f; lrow[tid] = 0.f; arow[tid] = 0.f; }
  __syncthreads();

  for (int e0 = 0; e0 < HW_; e0 += 32) {
    // Prefetch V fragments for this key tile (consumed after softmax barrier).
    v16bf vf[4];
#pragma unroll
    for (int j = 0; j < 4; ++j)
      vf[j] = make_b_frag(Vp, HW_, w * 64 + j * 16, e0, lane);

    // ---- S = (Q^T K) tile: partial over this wave's c-chunk ----
    v8f sacc[2][2];
#pragma unroll
    for (int i = 0; i < 2; ++i)
#pragma unroll
      for (int j = 0; j < 2; ++j)
#pragma unroll
        for (int r = 0; r < 8; ++r) sacc[i][j][r] = 0.f;
#pragma unroll
    for (int st = 0; st < 2; ++st) {
      int k_off = w * 64 + st * 32;
      v16bf kf[2];
#pragma unroll
      for (int j = 0; j < 2; ++j)
        kf[j] = make_b_frag(Kp, C_, e0 + j * 16, k_off, lane);
#pragma unroll
      for (int i = 0; i < 2; ++i)
#pragma unroll
        for (int j = 0; j < 2; ++j)
          sacc[i][j] = wmma_bf16(qf[i][st], kf[j], sacc[i][j]);
    }
#pragma unroll
    for (int i = 0; i < 2; ++i)
#pragma unroll
      for (int j = 0; j < 2; ++j)
#pragma unroll
        for (int r = 0; r < 8; ++r)
          Sp[w][(i * 16 + r + 8 * half) * 33 + j * 16 + nl] = sacc[i][j][r];
    __syncthreads();

    // ---- online softmax row update (thread r owns row r) ----
    if (tid < 32) {
      float sv[32];
      float mx = mrow[tid];
#pragma unroll
      for (int e = 0; e < 32; ++e) {
        float v = 0.f;
#pragma unroll
        for (int w8 = 0; w8 < 8; ++w8) v += Sp[w8][tid * 33 + e];
        sv[e] = v;
        mx = fmaxf(mx, v);
      }
      float al = __expf(mrow[tid] - mx);
      float sum = 0.f;
#pragma unroll
      for (int e = 0; e < 32; ++e) {
        float pv = __expf(sv[e] - mx);
        Ps[tid * PLD + e] = (bf16)pv;
        sum += pv;
      }
      lrow[tid] = lrow[tid] * al + sum;
      mrow[tid] = mx;
      arow[tid] = al;
    }
    __syncthreads();

    // ---- O = O*alpha + P * V^T, wave w owns c-chunk [w*64, w*64+64) ----
    v16bf pf[2];
#pragma unroll
    for (int i = 0; i < 2; ++i) pf[i] = make_a_frag(Ps, PLD, i * 16, 0, lane);
#pragma unroll
    for (int j = 0; j < 4; ++j) {
#pragma unroll
      for (int i = 0; i < 2; ++i) {
        int dbase = i * 16 + 8 * half;
        v8f t = oacc[i][j];
#pragma unroll
        for (int r = 0; r < 8; ++r) t[r] *= arow[dbase + r];
        oacc[i][j] = wmma_bf16(pf[i], vf[j], t);
      }
    }
  }

  // ---- finalize: divide by row sums, store [b][s][c] bf16 ----
  __syncthreads();
#pragma unroll
  for (int j = 0; j < 4; ++j)
#pragma unroll
    for (int i = 0; i < 2; ++i) {
      int dbase = i * 16 + 8 * half;
#pragma unroll
      for (int r = 0; r < 8; ++r) {
        int d = dbase + r;
        float v = oacc[i][j][r] / lrow[d];
        Og[(size_t)b * HW_ * C_ + (size_t)(q0 + d) * C_ + w * 64 + j * 16 + nl] = (bf16)v;
      }
    }
}

// ---------------------------------------------------------------------------
// Host launcher
// ---------------------------------------------------------------------------
extern "C" void kernel_launch(void* const* d_in, const int* in_sizes, int n_in,
                              void* d_out, int out_size, void* d_ws, size_t ws_size,
                              hipStream_t stream) {
  (void)in_sizes; (void)n_in; (void)out_size; (void)ws_size;
  const float* x   = (const float*)d_in[0];
  const float* nsc = (const float*)d_in[1];
  const float* nbi = (const float*)d_in[2];
  const float* wq  = (const float*)d_in[3];
  const float* bq  = (const float*)d_in[4];
  const float* wk  = (const float*)d_in[5];
  const float* bk  = (const float*)d_in[6];
  const float* wv  = (const float*)d_in[7];
  const float* bv  = (const float*)d_in[8];
  const float* wp  = (const float*)d_in[9];
  const float* bp  = (const float*)d_in[10];
  float* out = (float*)d_out;

  // Workspace carve-out (~164 MB total)
  uint8_t* ws = (uint8_t*)d_ws;
  const size_t WSZ = (size_t)512 * 512 * sizeof(bf16);   // 512 KB per weight
  bf16*  wqb   = (bf16*)(ws + 0 * WSZ);
  bf16*  wkb   = (bf16*)(ws + 1 * WSZ);
  bf16*  wvb   = (bf16*)(ws + 2 * WSZ);
  bf16*  wpb   = (bf16*)(ws + 3 * WSZ);
  float* stats = (float*)(ws + 4 * WSZ);                 // 2 KB
  bf16*  Hn = (bf16*)(ws + (4ull   << 20));              // 32 MB each
  bf16*  Qg = (bf16*)(ws + (36ull  << 20));
  bf16*  Kg = (bf16*)(ws + (68ull  << 20));
  bf16*  Vg = (bf16*)(ws + (100ull << 20));
  bf16*  Og = (bf16*)(ws + (132ull << 20));

  k_w2bf<<<dim3(1024), 256, 0, stream>>>(wq, wqb, 512 * 512);
  k_w2bf<<<dim3(1024), 256, 0, stream>>>(wk, wkb, 512 * 512);
  k_w2bf<<<dim3(1024), 256, 0, stream>>>(wv, wvb, 512 * 512);
  k_w2bf<<<dim3(1024), 256, 0, stream>>>(wp, wpb, 512 * 512);

  k_gnstats<<<dim3(B_ * NG), 256, 0, stream>>>(x, stats);
  k_gnapply<<<dim3(HW_ / 32, C_ / 32, B_), 256, 0, stream>>>(x, stats, nsc, nbi, Hn);

  const float qscale = 1.0f / sqrtf((float)C_);
  k_gemm<<<dim3(HW_ / 128, C_ / 128, B_), 256, 0, stream>>>(wqb, Hn, bq, nullptr, Qg, nullptr, 0, qscale);
  k_gemm<<<dim3(HW_ / 128, C_ / 128, B_), 256, 0, stream>>>(wkb, Hn, bk, nullptr, Kg, nullptr, 1, 1.0f);
  k_gemm<<<dim3(HW_ / 128, C_ / 128, B_), 256, 0, stream>>>(wvb, Hn, bv, nullptr, Vg, nullptr, 2, 1.0f);

  k_flash<<<dim3(HW_ / 32, B_), 256, 0, stream>>>(Qg, Kg, Vg, Og);

  k_gemm<<<dim3(HW_ / 128, C_ / 128, B_), 256, 0, stream>>>(wpb, Og, bp, x, nullptr, out, 3, 1.0f);
}
